// Qwen3MoeSparseMoeBlock_74217034875244
// MI455X (gfx1250) — matrix operand order repaired and checked by reading, not compiled, not run on hardware
//
#include <hip/hip_runtime.h>

#define HIDDEN 2048
#define INTER  768
#define NEXP   64
#define TOPK   8
#define MT     64              // token rows per expert tile
#define ZLD    (INTER + 8)     // LDS row stride (bf16 elems), 776*2B = 1552B (16B multiple)

typedef __attribute__((ext_vector_type(16))) __bf16 v16bf;
typedef __attribute__((ext_vector_type(8)))  float  v8f;

// A-matrix fragment (16x32 bf16): lane<16 holds row m, K = 0..15 of the
// 32-wide window; lane>=16 holds K = 16..31. p = &X[row][kk].
// fp32 -> bf16 via native cast => v_cvt_pk_bf16_f32 pairs.
__device__ __forceinline__ v16bf fragA_g(const float* p, int half) {
  const float4* q0 = (const float4*)(p + half * 16);
  const float4* q1 = (const float4*)(p + half * 16 + 8);
  float4 a = q0[0], b = q0[1], c = q1[0], d = q1[1];
  v16bf v;
  v[0]=(__bf16)a.x;  v[1]=(__bf16)a.y;  v[2]=(__bf16)a.z;  v[3]=(__bf16)a.w;
  v[4]=(__bf16)b.x;  v[5]=(__bf16)b.y;  v[6]=(__bf16)b.z;  v[7]=(__bf16)b.w;
  v[8]=(__bf16)c.x;  v[9]=(__bf16)c.y;  v[10]=(__bf16)c.z; v[11]=(__bf16)c.w;
  v[12]=(__bf16)d.x; v[13]=(__bf16)d.y; v[14]=(__bf16)d.z; v[15]=(__bf16)d.w;
  return v;
}

// B-matrix fragment (32x16 bf16): lane holds column n = lane&15, K = half*16..+15.
// B[k][n] = W[n][k] (weights are [out,in] row-major) -> 16 contiguous fp32.
__device__ __forceinline__ v16bf fragB_g(const float* p, int half) {
  const float4* q = (const float4*)(p + half * 16);
  float4 a = q[0], b = q[1], c = q[2], d = q[3];
  v16bf v;
  v[0]=(__bf16)a.x;  v[1]=(__bf16)a.y;  v[2]=(__bf16)a.z;  v[3]=(__bf16)a.w;
  v[4]=(__bf16)b.x;  v[5]=(__bf16)b.y;  v[6]=(__bf16)b.z;  v[7]=(__bf16)b.w;
  v[8]=(__bf16)c.x;  v[9]=(__bf16)c.y;  v[10]=(__bf16)c.z; v[11]=(__bf16)c.w;
  v[12]=(__bf16)d.x; v[13]=(__bf16)d.y; v[14]=(__bf16)d.z; v[15]=(__bf16)d.w;
  return v;
}

// A fragment from bf16 LDS (two 16B ds_load_b128 per lane)
__device__ __forceinline__ v16bf fragA_lds(const __bf16* p, int half) {
  union { uint4 q[2]; v16bf v; } u;
  u.q[0] = *(const uint4*)(p + half * 16);
  u.q[1] = *(const uint4*)(p + half * 16 + 8);
  return u.v;
}

// ---------------- kernel 0: zero output + expert counts ----------------
__global__ void moe_zero_kernel(float* __restrict__ out, int n, int* __restrict__ counts) {
  int i = blockIdx.x * blockDim.x + threadIdx.x;
  if (i < n) out[i] = 0.0f;
  if (i < NEXP) counts[i] = 0;
}

// ---------------- kernel 1: router (softmax + top-8 + scatter) ----------------
__global__ __launch_bounds__(64) void moe_router_kernel(
    const float* __restrict__ x, const float* __restrict__ gate_w,
    int* __restrict__ counts, int* __restrict__ tlist, float* __restrict__ wlist,
    int T) {
  __shared__ float xs[HIDDEN];
  __shared__ float logit[NEXP];
  const int t   = blockIdx.x;
  const int tid = threadIdx.x;

  for (int i = tid; i < HIDDEN; i += 64) xs[i] = x[(size_t)t * HIDDEN + i];
  __syncthreads();

  const float* gw = gate_w + (size_t)tid * HIDDEN;
  float acc = 0.0f;
#pragma unroll 8
  for (int i = 0; i < HIDDEN; ++i) acc += xs[i] * gw[i];
  logit[tid] = acc;
  __syncthreads();

  if (tid == 0) {
    float mx = -1e30f;
    for (int e = 0; e < NEXP; ++e) mx = fmaxf(mx, logit[e]);
    float p[NEXP];
    for (int e = 0; e < NEXP; ++e) p[e] = expf(logit[e] - mx);
    // top-8 (unnormalized exp; renormalizing over the subset == reference)
    int   sel[TOPK]; float selw[TOPK]; float wsum = 0.0f;
    bool  used[NEXP];
    for (int e = 0; e < NEXP; ++e) used[e] = false;
    for (int k = 0; k < TOPK; ++k) {
      int best = 0; float bv = -1.0f;
      for (int e = 0; e < NEXP; ++e)
        if (!used[e] && p[e] > bv) { bv = p[e]; best = e; }
      used[best] = true; sel[k] = best; selw[k] = bv; wsum += bv;
    }
    float inv = 1.0f / wsum;
    for (int k = 0; k < TOPK; ++k) {
      int e = sel[k];
      int slot = atomicAdd(&counts[e], 1);
      tlist[(size_t)e * T + slot] = t;
      wlist[(size_t)e * T + slot] = selw[k] * inv;
    }
  }
}

// ---------------- kernel 2: fused expert SwiGLU + down-proj ----------------
// grid.x = expert, grid.y = 64-token tile; 256 threads = 8 waves.
// wave -> (m_sub = wave&3) x (n group = (wave>>2)*64); 4 n-subtiles per wave.
__global__ __launch_bounds__(256) void moe_expert_kernel(
    const float* __restrict__ x,  const float* __restrict__ Wg,
    const float* __restrict__ Wu, const float* __restrict__ Wd,
    const int* __restrict__ counts, const int* __restrict__ tlist,
    const float* __restrict__ wlist, float* __restrict__ out, int T) {
  extern __shared__ char smem[];
  __bf16* zs  = (__bf16*)smem;                                       // [MT][ZLD] bf16
  int*    tok = (int*)(smem + (size_t)MT * ZLD * 2);                 // [MT]
  float*  wrw = (float*)(smem + (size_t)MT * ZLD * 2 + MT * 4);      // [MT]

  const int e   = blockIdx.x;
  const int cnt = counts[e];
  const int m0  = blockIdx.y * MT;
  if (m0 >= cnt) return;

  const int tid = threadIdx.x;
  if (tid < MT) {
    int slot = m0 + tid;
    if (slot < cnt) { tok[tid] = tlist[(size_t)e * T + slot]; wrw[tid] = wlist[(size_t)e * T + slot]; }
    else            { tok[tid] = 0;                           wrw[tid] = 0.0f; }
  }
  __syncthreads();

  const int wave  = tid >> 5;
  const int lane  = tid & 31;
  const int half  = lane >> 4;
  const int l16   = lane & 15;
  const int m_sub = wave & 3;
  const int n_grp = (wave >> 2) * 64;

  const float* Wg_e = Wg + (size_t)e * INTER * HIDDEN;
  const float* Wu_e = Wu + (size_t)e * INTER * HIDDEN;
  const float* Wd_e = Wd + (size_t)e * HIDDEN * INTER;

  // per-lane row metadata for accumulator rows (C/D layout: row = v + 8*half)
  float wv[8]; int tokv[8];
#pragma unroll
  for (int v = 0; v < 8; ++v) {
    int row = m_sub * 16 + v + 8 * half;
    wv[v] = wrw[row]; tokv[v] = tok[row];
  }
  const float* xrow = x + (size_t)tok[m_sub * 16 + l16] * HIDDEN;

  // ---- Phase 1: g,u = X@Wg^T, X@Wu^T over I, z = silu(g)*u*w -> LDS (bf16) ----
  for (int nt = 0; nt < INTER / 128; ++nt) {
    v8f accg[4] = {}, accu[4] = {};
    const int nb = nt * 128 + n_grp;
    const float* wgrow[4]; const float* wurow[4];
#pragma unroll
    for (int j = 0; j < 4; ++j) {
      int n = nb + j * 16 + l16;
      wgrow[j] = Wg_e + (size_t)n * HIDDEN;
      wurow[j] = Wu_e + (size_t)n * HIDDEN;
    }
    for (int kk = 0; kk < HIDDEN; kk += 32) {
      v16bf a = fragA_g(xrow + kk, half);
#pragma unroll
      for (int j = 0; j < 4; ++j) {
        v16bf bg = fragB_g(wgrow[j] + kk, half);
        accg[j] = __builtin_amdgcn_wmma_f32_16x16x32_bf16(false, a, false, bg, (short)0, accg[j], false, false);
        v16bf bu = fragB_g(wurow[j] + kk, half);
        accu[j] = __builtin_amdgcn_wmma_f32_16x16x32_bf16(false, a, false, bu, (short)0, accu[j], false, false);
      }
    }
#pragma unroll
    for (int j = 0; j < 4; ++j) {
      int n = nb + j * 16 + l16;
#pragma unroll
      for (int v = 0; v < 8; ++v) {
        int   row = m_sub * 16 + v + 8 * half;
        float g = accg[j][v], u = accu[j][v];
        float z = (g / (1.0f + __expf(-g))) * u * wv[v];
        zs[(size_t)row * ZLD + n] = (__bf16)z;
      }
    }
  }
  __syncthreads();

  // ---- Phase 2: out[t,:] += z @ Wd^T (atomic scatter) ----
  const __bf16* zrow = zs + (size_t)(m_sub * 16 + l16) * ZLD;
  for (int ht = 0; ht < HIDDEN / 128; ++ht) {
    v8f acc[4] = {};
    const int hb = ht * 128 + n_grp;
    const float* wdrow[4];
#pragma unroll
    for (int j = 0; j < 4; ++j)
      wdrow[j] = Wd_e + (size_t)(hb + j * 16 + l16) * INTER;
    for (int kk = 0; kk < INTER; kk += 32) {
      v16bf a = fragA_lds(zrow + kk, half);
#pragma unroll
      for (int j = 0; j < 4; ++j) {
        v16bf b = fragB_g(wdrow[j] + kk, half);
        acc[j] = __builtin_amdgcn_wmma_f32_16x16x32_bf16(false, a, false, b, (short)0, acc[j], false, false);
      }
    }
#pragma unroll
    for (int j = 0; j < 4; ++j) {
      int h = hb + j * 16 + l16;
#pragma unroll
      for (int v = 0; v < 8; ++v)
        atomicAdd(&out[(size_t)tokv[v] * HIDDEN + h], acc[j][v]);
    }
  }
}

extern "C" void kernel_launch(void* const* d_in, const int* in_sizes, int n_in,
                              void* d_out, int out_size, void* d_ws, size_t ws_size,
                              hipStream_t stream) {
  const float* x      = (const float*)d_in[0];
  const float* gate_w = (const float*)d_in[1];
  const float* Wg     = (const float*)d_in[2];
  const float* Wu     = (const float*)d_in[3];
  const float* Wd     = (const float*)d_in[4];
  float* out = (float*)d_out;

  const int T = in_sizes[0] / HIDDEN;   // 1024 tokens

  int*   counts = (int*)d_ws;
  int*   tlist  = counts + NEXP;
  float* wlist  = (float*)(tlist + (size_t)NEXP * T);

  const int nOut = T * HIDDEN;
  moe_zero_kernel<<<(nOut + 255) / 256, 256, 0, stream>>>(out, nOut, counts);
  moe_router_kernel<<<T, 64, 0, stream>>>(x, gate_w, counts, tlist, wlist, T);

  size_t shmem = (size_t)MT * ZLD * 2 + (size_t)MT * 4 + (size_t)MT * 4; // ~99.8 KB
  dim3 grid(NEXP, (T + MT - 1) / MT);
  moe_expert_kernel<<<grid, 256, shmem, stream>>>(x, Wg, Wu, Wd, counts, tlist, wlist, out, T);
}
